// GruNet_44315472560536
// MI455X (gfx1250) — compile-verified, matrix operand order checked by reading, not run on hardware
//
#include <hip/hip_runtime.h>
#include <hip/hip_bf16.h>
#include <cmath>

// ---------------------------------------------------------------------------
// GRU network (SEQ=512, BATCH=64, VOCAB=HID=1024, LAYERS=3) on gfx1250.
// All GEMMs via v_wmma_f32_16x16x32_bf16, B-operands staged through LDS with
// a double-buffered pipeline (async global->LDS copies when available).
// ---------------------------------------------------------------------------

typedef __bf16 bf16_t;
typedef __attribute__((ext_vector_type(16))) __bf16 v16bf;
typedef __attribute__((ext_vector_type(8)))  __bf16 v8bf;
typedef __attribute__((ext_vector_type(8)))  float  v8f;
typedef __attribute__((ext_vector_type(4)))  int    v4i;

#define WMMA_BF16(a, b, c) \
    __builtin_amdgcn_wmma_f32_16x16x32_bf16(false, (a), false, (b), (short)0, (c), false, false)

#if defined(__has_builtin)
#  if __has_builtin(__builtin_amdgcn_global_load_async_to_lds_b128)
#    define USE_ASYNC_LDS 1
#  endif
#endif

// 16-byte global -> LDS copy (async if the gfx1250 builtin exists).
__device__ __forceinline__ void copy16_g2lds(const bf16_t* g, bf16_t* l) {
#ifdef USE_ASYNC_LDS
    __builtin_amdgcn_global_load_async_to_lds_b128(
        (v4i*)(void*)g,
        (__attribute__((address_space(3))) v4i*)l, 0, 0);
#else
    *(v8bf*)l = *(const v8bf*)g;
#endif
}

// Wait for my async stage traffic, then block barrier.
__device__ __forceinline__ void stage_barrier() {
#if defined(USE_ASYNC_LDS) && defined(__AMDGCN__)
    asm volatile("s_wait_asynccnt 0x0" ::: "memory");
#endif
    __syncthreads();
}

// Load one 16x32 bf16 operand tile from a row-major matrix (leading dim `ld`).
// ISA layout: lanes 0-15 -> row = lane, K = {0..7,16..23};
// lanes 16-31 -> row = lane-16, K = {8..15,24..31}.  Two 16B loads per lane.
__device__ __forceinline__ v16bf load_tile16x32(const bf16_t* __restrict__ base,
                                                int rowBase, int ld, int kBase, int lane) {
    const int r    = rowBase + (lane & 15);
    const int koff = (lane >> 4) ? 8 : 0;
    const bf16_t* p = base + (size_t)r * ld + kBase + koff;
    v8bf lo = *(const v8bf*)(p);
    v8bf hi = *(const v8bf*)(p + 16);
    v16bf out;
#pragma unroll
    for (int i = 0; i < 8; ++i) { out[i] = lo[i]; out[i + 8] = hi[i]; }
    return out;
}

// Same tile fetch but from an LDS stripe with padded pitch (40 bf16 = 80 B).
constexpr int LPITCH = 40;
__device__ __forceinline__ v16bf lds_tile16x32(const bf16_t* smemRows, int rowBase, int lane) {
    const int r    = rowBase + (lane & 15);
    const int koff = (lane >> 4) ? 8 : 0;
    const bf16_t* p = smemRows + r * LPITCH + koff;
    v8bf lo = *(const v8bf*)(p);
    v8bf hi = *(const v8bf*)(p + 16);
    v16bf out;
#pragma unroll
    for (int i = 0; i < 8; ++i) { out[i] = lo[i]; out[i + 8] = hi[i]; }
    return out;
}

// ---------------------------------------------------------------------------
// f32 -> bf16 conversion (sizes here are all multiples of 4)
// ---------------------------------------------------------------------------
__global__ void __launch_bounds__(256) cvt_f32_to_bf16(const float* __restrict__ in,
                                                       bf16_t* __restrict__ out, long n4) {
    long i = ((long)blockIdx.x * blockDim.x + threadIdx.x) * 4;
    if (i < n4) {
        float4 v = *(const float4*)(in + i);
        out[i + 0] = (bf16_t)v.x; out[i + 1] = (bf16_t)v.y;
        out[i + 2] = (bf16_t)v.z; out[i + 3] = (bf16_t)v.w;
    }
}

__global__ void __launch_bounds__(256) fill_zero_bf16(bf16_t* __restrict__ p, int n) {
    int i = blockIdx.x * blockDim.x + threadIdx.x;
    if (i < n) p[i] = (bf16_t)0.0f;
}

// ---------------------------------------------------------------------------
// GEMM: out[M,N] = A[M,K](bf16) * W[N,K](bf16)^T (+ bias), WMMA bf16.
// Block = 128 thr (4 waves), block tile 64M x 64N.  Wave w owns rows
// mBase = by*64 + w*16 and all four 16-col subtiles (4 accumulators reusing
// one A tile).  The 64x32 B chunk is staged in LDS (shared by all 4 waves),
// double-buffered: stage chunk i+1 (async) while WMMAs consume chunk i.
// ---------------------------------------------------------------------------
template <bool BF16OUT>
__device__ __forceinline__ void gemm64x64_body(
    const bf16_t* __restrict__ A, const bf16_t* __restrict__ W,
    const float* __restrict__ bias, void* __restrict__ outp,
    int M, int N, int K)
{
    __shared__ bf16_t smem[2][64 * LPITCH];

    const int tid   = threadIdx.x;
    const int lane  = tid & 31;
    const int wave  = tid >> 5;
    const int nBase = blockIdx.x * 64;
    const int mBase = blockIdx.y * 64 + wave * 16;

    // Stage 64 rows x 32 cols of W (k-chunk `kb`) into smem[buf].
    auto stage = [&](int buf, int kb) {
#pragma unroll
        for (int s = tid; s < 256; s += 128) {          // 256 x 16B segments
            const int row = s >> 2, seg = s & 3;
            copy16_g2lds(W + (size_t)(nBase + row) * K + kb + seg * 8,
                         &smem[buf][row * LPITCH + seg * 8]);
        }
    };

    v8f acc0 = {}, acc1 = {}, acc2 = {}, acc3 = {};
    const int nk = K / 32;
    stage(0, 0);
    for (int i = 0; i < nk; ++i) {
        stage_barrier();                                // chunk i resident
        if (i + 1 < nk) stage((i + 1) & 1, (i + 1) * 32);
        const bf16_t* bufp = smem[i & 1];
        v16bf a  = load_tile16x32(A, mBase, K, i * 32, lane);
        v16bf b0 = lds_tile16x32(bufp,  0, lane);
        v16bf b1 = lds_tile16x32(bufp, 16, lane);
        v16bf b2 = lds_tile16x32(bufp, 32, lane);
        v16bf b3 = lds_tile16x32(bufp, 48, lane);
        acc0 = WMMA_BF16(a, b0, acc0);
        acc1 = WMMA_BF16(a, b1, acc1);
        acc2 = WMMA_BF16(a, b2, acc2);
        acc3 = WMMA_BF16(a, b3, acc3);
    }

    // C/D layout: vgpr i, lanes 0-15 -> M=i; lanes 16-31 -> M=8+i; N = lane%16.
    const int rowBase = mBase + ((lane >> 4) << 3);
    v8f* accs[4] = { &acc0, &acc1, &acc2, &acc3 };
#pragma unroll
    for (int j = 0; j < 4; ++j) {
        const int col = nBase + j * 16 + (lane & 15);
        const float bv = bias ? bias[col] : 0.0f;
#pragma unroll
        for (int i = 0; i < 8; ++i) {
            const float v = (*accs[j])[i] + bv;
            if (BF16OUT) ((bf16_t*)outp)[(size_t)(rowBase + i) * N + col] = (bf16_t)v;
            else         ((float*)outp)[(size_t)(rowBase + i) * N + col] = v;
        }
    }
}

__global__ void __launch_bounds__(128) gemm_bf16_bias_bf16out(
    const bf16_t* __restrict__ A, const bf16_t* __restrict__ W,
    const float* __restrict__ bias, bf16_t* __restrict__ out, int M, int N, int K) {
    gemm64x64_body<true>(A, W, bias, out, M, N, K);
}

__global__ void __launch_bounds__(128) gemm_bf16_f32out(
    const bf16_t* __restrict__ A, const bf16_t* __restrict__ W,
    float* __restrict__ out, int M, int N, int K) {
    gemm64x64_body<false>(A, W, nullptr, out, M, N, K);
}

// ---------------------------------------------------------------------------
// One GRU timestep:  hg = h_in * W_hh^T + b_hh ; gates ; h_out/ys = h_new.
// Grid = 64 blocks (one per 16-col group of H) x 128 thr (4 waves = 4 batch
// tiles).  All waves share the same r/z/n W_hh tiles -> staged once in LDS
// (48 rows x 32 k per chunk, double-buffered, async).  Gate math block-local.
// ---------------------------------------------------------------------------
__global__ void __launch_bounds__(128) gru_step_kernel(
    const bf16_t* __restrict__ h_in,   // [64, 1024]
    bf16_t*       __restrict__ h_out,  // [64, 1024]
    const bf16_t* __restrict__ xg_t,   // [64, 3072]  (b_ih pre-added)
    const bf16_t* __restrict__ Whh,    // [3072, 1024]
    const float*  __restrict__ bhh,    // [3072]
    bf16_t*       __restrict__ ys_t)   // [64, 1024]
{
    constexpr int H = 1024, G3 = 3 * 1024;
    __shared__ bf16_t smem[2][48 * LPITCH];

    const int tid   = threadIdx.x;
    const int lane  = tid & 31;
    const int wave  = tid >> 5;
    const int mBase = wave * 16;           // batch rows
    const int cBase = blockIdx.x * 16;     // hidden columns

    // Stage the 3 gate tiles (rows: gate*H + cBase + rr) for k-chunk `kb`.
    auto stage = [&](int buf, int kb) {
        for (int s = tid; s < 192; s += 128) {          // 48 rows x 4 segs
            const int rIdx = s >> 2, seg = s & 3;
            const int gate = rIdx >> 4, rr = rIdx & 15;
            copy16_g2lds(Whh + (size_t)(gate * H + cBase + rr) * H + kb + seg * 8,
                         &smem[buf][rIdx * LPITCH + seg * 8]);
        }
    };

    v8f accR = {}, accZ = {}, accN = {};
    const int nk = H / 32;
    stage(0, 0);
    for (int i = 0; i < nk; ++i) {
        stage_barrier();
        if (i + 1 < nk) stage((i + 1) & 1, (i + 1) * 32);
        const bf16_t* bufp = smem[i & 1];
        v16bf a  = load_tile16x32(h_in, mBase, H, i * 32, lane);
        v16bf br = lds_tile16x32(bufp,  0, lane);
        v16bf bz = lds_tile16x32(bufp, 16, lane);
        v16bf bn = lds_tile16x32(bufp, 32, lane);
        accR = WMMA_BF16(a, br, accR);
        accZ = WMMA_BF16(a, bz, accZ);
        accN = WMMA_BF16(a, bn, accN);
    }

    const int col     = cBase + (lane & 15);
    const int rowBase = mBase + ((lane >> 4) << 3);
    const float br_b = bhh[col];
    const float bz_b = bhh[H + col];
    const float bn_b = bhh[2 * H + col];

#pragma unroll
    for (int i = 0; i < 8; ++i) {
        const int row = rowBase + i;
        const float xr = (float)xg_t[(size_t)row * G3 + col];
        const float xz = (float)xg_t[(size_t)row * G3 + H + col];
        const float xn = (float)xg_t[(size_t)row * G3 + 2 * H + col];
        const float hr = accR[i] + br_b;
        const float hz = accZ[i] + bz_b;
        const float hn = accN[i] + bn_b;
        const float r = 1.0f / (1.0f + __expf(-(xr + hr)));
        const float z = 1.0f / (1.0f + __expf(-(xz + hz)));
        const float n = tanhf(xn + r * hn);
        const float hp = (float)h_in[(size_t)row * H + col];
        const float hnew = (1.0f - z) * n + z * hp;
        const bf16_t hb = (bf16_t)hnew;
        h_out[(size_t)row * H + col] = hb;
        ys_t [(size_t)row * H + col] = hb;
    }
}

// ---------------------------------------------------------------------------
// Host orchestration
// ---------------------------------------------------------------------------
extern "C" void kernel_launch(void* const* d_in, const int* in_sizes, int n_in,
                              void* d_out, int out_size, void* d_ws, size_t ws_size,
                              hipStream_t stream) {
    (void)in_sizes; (void)n_in; (void)out_size; (void)ws_size;

    constexpr int T = 512, B = 64, H = 1024, V = 1024, L = 3;
    constexpr long M  = (long)T * B;      // 32768 rows
    constexpr long G3 = 3L * H;           // 3072

    const float* x     = (const float*)d_in[0];
    const float* W_ih  = (const float*)d_in[1];
    const float* W_hh  = (const float*)d_in[2];
    const float* b_ih  = (const float*)d_in[3];
    const float* b_hh  = (const float*)d_in[4];
    const float* W_out = (const float*)d_in[5];
    float* out = (float*)d_out;

    // Workspace carve-up (bf16 staging), 256B aligned.
    char* ws = (char*)d_ws;
    size_t off = 0;
    auto carve = [&](size_t bytes) {
        void* p = ws + off;
        off = (off + bytes + 255) & ~(size_t)255;
        return p;
    };
    bf16_t* wih_b  = (bf16_t*)carve((size_t)L * G3 * H * sizeof(bf16_t)); // 18.9 MB
    bf16_t* whh_b  = (bf16_t*)carve((size_t)L * G3 * H * sizeof(bf16_t)); // 18.9 MB
    bf16_t* wout_b = (bf16_t*)carve((size_t)V * H * sizeof(bf16_t));      //  2.1 MB
    bf16_t* cur    = (bf16_t*)carve((size_t)M * H * sizeof(bf16_t));      // 67.1 MB
    bf16_t* xg     = (bf16_t*)carve((size_t)M * G3 * sizeof(bf16_t));     // 201 MB
    bf16_t* h0     = (bf16_t*)carve((size_t)B * H * sizeof(bf16_t));
    bf16_t* h1     = (bf16_t*)carve((size_t)B * H * sizeof(bf16_t));

    auto cvt = [&](const float* src, bf16_t* dst, long n) {
        long nth = n / 4;
        cvt_f32_to_bf16<<<(unsigned)((nth + 255) / 256), 256, 0, stream>>>(src, dst, n);
    };
    cvt(x,     cur,    M * H);
    cvt(W_ih,  wih_b,  (long)L * G3 * H);
    cvt(W_hh,  whh_b,  (long)L * G3 * H);
    cvt(W_out, wout_b, (long)V * H);

    const dim3 blk(128);
    for (int l = 0; l < L; ++l) {
        const bf16_t* wih_l = wih_b + (size_t)l * G3 * H;
        const bf16_t* whh_l = whh_b + (size_t)l * G3 * H;
        const float*  bih_l = b_ih + (size_t)l * G3;
        const float*  bhh_l = b_hh + (size_t)l * G3;

        // xg = cur * W_ih^T + b_ih   -> [M, 3H] bf16
        gemm_bf16_bias_bf16out<<<dim3(G3 / 64, M / 64), blk, 0, stream>>>(
            cur, wih_l, bih_l, xg, (int)M, (int)G3, H);

        // h0 = 0
        fill_zero_bf16<<<(B * H + 255) / 256, 256, 0, stream>>>(h0, B * H);

        // Sequential recurrence (double-buffered h).
        bf16_t* hin = h0; bf16_t* hout = h1;
        for (int t = 0; t < T; ++t) {
            gru_step_kernel<<<dim3(H / 16), blk, 0, stream>>>(
                hin, hout,
                xg + (size_t)t * B * G3,
                whh_l, bhh_l,
                cur + (size_t)t * B * H);   // ys overwrites layer input in place
            bf16_t* tmp = hin; hin = hout; hout = tmp;
        }
        // `cur` now holds this layer's output sequence.
    }

    // out = ys * W_out^T  -> [M, V] f32
    gemm_bf16_f32out<<<dim3(V / 64, M / 64), blk, 0, stream>>>(
        cur, wout_b, out, (int)M, V, H);
}
